// TiLayer_16063177687661
// MI455X (gfx1250) — matrix operand order repaired and checked by reading, not compiled, not run on hardware
//
#include <hip/hip_runtime.h>

#define N_TOK 4096
#define D_DIM 64
#define H_DIM 256
#define V_DIM 8192
#define EPSLN 1e-6f
#define LOG_ALPHA 3.5f

typedef __attribute__((ext_vector_type(16))) _Float16 v16h;
typedef __attribute__((ext_vector_type(8)))  _Float16 v8h;
typedef __attribute__((ext_vector_type(8)))  float    v8f;
typedef __attribute__((ext_vector_type(4)))  float    v4f;

#define DEV static __device__ __forceinline__

// ---- WMMA fragment loaders (CDNA5 16x16x32 f16 layouts, ISA 7.12.2) ----
// A (16x32 f16, MxK) row-major with stride lda (multiple of 8):
// lanes 0-15: M=lane, K in {0..7}u{16..23}; lanes 16-31: M=lane-16, K in
// {8..15}u{24..31}.  Two aligned 16B chunks -> 2x ds_load_b128.
DEV v16h frag_A(const _Float16* A, int lane, int k0, int lda) {
  int m  = lane & 15;
  int kb = (lane >> 4) * 8;
  const v8h* p0 = (const v8h*)(A + m * lda + k0 + kb);
  const v8h* p1 = (const v8h*)(A + m * lda + k0 + 16 + kb);
  v8h lo = *p0, hi = *p1;
  return __builtin_shufflevector(lo, hi, 0, 1, 2, 3, 4, 5, 6, 7,
                                 8, 9, 10, 11, 12, 13, 14, 15);
}

// B (32x16 f16, KxN) stored TRANSPOSED (N-major: Bt[n][k], stride ldb):
// lane n=lane&15 is column; lanes 0-15: K in {0..7}u{16..23}; lanes 16-31:
// K in {8..15}u{24..31} (matches frag_A) -> 2 aligned 16B loads.
DEV v16h frag_Bt(const _Float16* Bt, int lane, int k0, int ldb) {
  int n  = lane & 15;
  int kb = (lane >> 4) * 8;
  const v8h* p0 = (const v8h*)(Bt + (size_t)n * ldb + k0 + kb);
  const v8h* p1 = (const v8h*)(Bt + (size_t)n * ldb + k0 + 16 + kb);
  v8h lo = *p0, hi = *p1;
  return __builtin_shufflevector(lo, hi, 0, 1, 2, 3, 4, 5, 6, 7,
                                 8, 9, 10, 11, 12, 13, 14, 15);
}

DEV v8f wmma_f16(v16h a, v16h b, v8f c) {
  return __builtin_amdgcn_wmma_f32_16x16x32_f16(false, a, false, b,
                                                (short)0, c, false, false);
}

// C/D (16x16 f32): element j of v8f for lane l is row (l>=16?8:0)+j, col l&15.

// ---------------- K0: convert weights to f16, B-side transposed ------------
__global__ void k_convert(const float* outw, const float* w1, const float* w2,
                          _Float16* outhT, _Float16* w1T, _Float16* w2T) {
  int i = blockIdx.x * 256 + threadIdx.x;
  if (i < D_DIM * V_DIM) {           // outhT[v][d] = output[d][v]
    int v = i >> 6, d = i & 63;
    outhT[i] = (_Float16)outw[(size_t)d * V_DIM + v];
  }
  if (i < D_DIM * H_DIM) {           // w1T[h][d] = w1[d][h]
    int h = i >> 6, d = i & 63;
    w1T[i] = (_Float16)w1[d * H_DIM + h];
  }
  if (i < H_DIM * D_DIM) {           // w2T[d][h] = w2[h][d]
    int d = i >> 8, h = i & 255;
    w2T[i] = (_Float16)w2[h * D_DIM + d];
  }
}

// ---------------- K1: t = log_sigmoid(xcat @ delta); RNG divider drop ------
__global__ void k_time(const float* xcat, const float* delta, const int* xdiv,
                       const int* seedp, float* t, int* kept) {
  int i = blockIdx.x * blockDim.x + threadIdx.x;
  if (i >= N_TOK) return;
  const v4f* row = (const v4f*)(xcat + (size_t)i * D_DIM);
  const v4f* dl  = (const v4f*)delta;
  float z = 0.f;
#pragma unroll
  for (int d = 0; d < D_DIM / 4; ++d) {
    v4f a = row[d], b = dl[d];
    z += a.x * b.x + a.y * b.y + a.z * b.z + a.w * b.w;
  }
  t[i] = fminf(z, 0.f) - log1pf(__expf(-fabsf(z)));
  unsigned long long s = (unsigned long long)(unsigned)seedp[0] * 0x9E3779B97F4A7C15ULL
                       + (unsigned long long)i * 0xBF58476D1CE4E5B9ULL
                       + 0x94D049BB133111EBULL;
  s ^= s >> 30; s *= 0xBF58476D1CE4E5B9ULL;
  s ^= s >> 27; s *= 0x94D049BB133111EBULL;
  s ^= s >> 31;
  float u = (float)(s >> 40) * (1.0f / 16777216.0f);
  kept[i] = (u > 0.1f) ? xdiv[i] : 0;
}

// ---------------- K1b: cumsum of kept dividers -> segment key --------------
__global__ void k_divscan(const int* kept, const int* xsep, int* key) {
  __shared__ int ps[1024];
  int tid = threadIdx.x;
  int base = tid * 4;
  int l0 = kept[base];
  int l1 = l0 + kept[base + 1];
  int l2 = l1 + kept[base + 2];
  int l3 = l2 + kept[base + 3];
  ps[tid] = l3;
  __syncthreads();
  for (int off = 1; off < 1024; off <<= 1) {
    int v = (tid >= off) ? ps[tid - off] : 0;
    __syncthreads();
    ps[tid] += v;
    __syncthreads();
  }
  int excl = ps[tid] - l3;
  key[base]     = ((excl + l0) << 4) | (xsep[base] & 15);
  key[base + 1] = ((excl + l1) << 4) | (xsep[base + 1] & 15);
  key[base + 2] = ((excl + l2) << 4) | (xsep[base + 2] & 15);
  key[base + 3] = ((excl + l3) << 4) | (xsep[base + 3] & 15);
}

// ---------------- K2: sequential reverse recurrence ------------------------
__global__ void k_scan(const float* xcat, const float* t, const int* key,
                       const float* a_p, const float* b_p, float* xpre) {
  __shared__ float bp[D_DIM * D_DIM];
  __shared__ float w[D_DIM];
  int tid = threadIdx.x;  // 64 threads = 2 waves
  for (int e = tid; e < D_DIM * D_DIM; e += D_DIM) bp[e] = b_p[e];
  float ap = a_p[tid];
  float s  = xcat[(N_TOK - 1) * D_DIM + tid];
  xpre[(N_TOK - 1) * D_DIM + tid] = s;
  __syncthreads();
  for (int i = N_TOK - 2; i >= 0; --i) {
    float dt = t[i + 1] - t[i];
    bool  m  = (key[i] == key[i + 1]);
    w[tid] = s * __expf(dt * ap);
    __syncthreads();
    float y = xcat[i * D_DIM + tid];
    if (m) {
      float acc = 0.f;
#pragma unroll 8
      for (int d = 0; d < D_DIM; ++d) acc += w[d] * bp[d * D_DIM + tid];
      y += acc;
    }
    __syncthreads();
    s = y;
    xpre[i * D_DIM + tid] = y;
  }
}

// ---------------- K3/K5: column LayerNorm stats (token axis) ---------------
__global__ void k_colstats(const float* x, float* mu, float* rsig) {
  __shared__ float s1[256], s2[256];
  int d = blockIdx.x, tid = threadIdx.x;
  float a = 0.f, b = 0.f;
  for (int i = tid; i < N_TOK; i += 256) {
    float v = x[i * D_DIM + d];
    a += v; b += v * v;
  }
  s1[tid] = a; s2[tid] = b;
  __syncthreads();
  for (int off = 128; off; off >>= 1) {
    if (tid < off) { s1[tid] += s1[tid + off]; s2[tid] += s2[tid + off]; }
    __syncthreads();
  }
  if (tid == 0) {
    float m = s1[0] / (float)N_TOK;
    float var = s2[0] / (float)N_TOK - m * m;
    mu[d] = m;
    rsig[d] = rsqrtf(var + EPSLN);
  }
}

// ---------------- K4: WMMA FFN + residual, writes xnew into d_out ----------
__global__ void k_ffn(const float* xpre, const float* xcat, const float* mu,
                      const float* rsig, const _Float16* w1T, const float* b1,
                      const _Float16* w2T, const float* b2, float* xnew) {
  __shared__ __align__(32) _Float16 Ah[64 * 64];
  __shared__ __align__(32) _Float16 H1[64 * 256];
  __shared__ float b1s[256];
  int tid = threadIdx.x;                 // 256 threads = 8 waves
  int r0 = blockIdx.x * 64;
  b1s[tid] = b1[tid];
  for (int e = tid; e < 64 * 64; e += 256) {
    int row = e >> 6, d = e & 63;
    Ah[e] = (_Float16)((xpre[(r0 + row) * D_DIM + d] - mu[d]) * rsig[d]);
  }
  __syncthreads();
  int lane = tid & 31, wv = tid >> 5;
  int rt = wv >> 1;                      // each wave owns one row-tile
  // GEMM1: [64,64] @ [64,256] -> relu -> H1. Hoisted A fragments.
  {
    const _Float16* Ap = Ah + rt * 16 * 64;
    v16h a0 = frag_A(Ap, lane, 0, 64);
    v16h a1 = frag_A(Ap, lane, 32, 64);
    int col16 = lane & 15;
    int rbase = rt * 16 + ((lane >> 4) * 8);
    for (int tt = 0; tt < 8; ++tt) {
      int ct = (wv & 1) * 8 + tt;
      const _Float16* Bp = w1T + (size_t)(ct * 16) * 64;
      v8f c = {};
      c = wmma_f16(a0, frag_Bt(Bp, lane, 0, 64), c);
      c = wmma_f16(a1, frag_Bt(Bp, lane, 32, 64), c);
      int col = ct * 16 + col16;
#pragma unroll
      for (int j = 0; j < 8; ++j)
        H1[(rbase + j) * 256 + col] = (_Float16)fmaxf(c[j] + b1s[col], 0.f);
    }
  }
  __syncthreads();
  // GEMM2: [64,256] @ [256,64] + b2 + xcat. Two col-tiles per wave, shared A.
  {
    const _Float16* Ap = H1 + rt * 16 * 256;
    int ct0 = (wv & 1) * 2;
    v8f c0 = {}, c1 = {};
#pragma unroll
    for (int ks = 0; ks < H_DIM; ks += 32) {
      v16h a = frag_A(Ap, lane, ks, 256);
      c0 = wmma_f16(a, frag_Bt(w2T + (size_t)(ct0 * 16) * 256, lane, ks, 256), c0);
      c1 = wmma_f16(a, frag_Bt(w2T + (size_t)((ct0 + 1) * 16) * 256, lane, ks, 256), c1);
    }
    int rbase = rt * 16 + ((lane >> 4) * 8);
#pragma unroll
    for (int tt = 0; tt < 2; ++tt) {
      v8f c = tt ? c1 : c0;
      int col = (ct0 + tt) * 16 + (lane & 15);
#pragma unroll
      for (int j = 0; j < 8; ++j) {
        int row = r0 + rbase + j;
        xnew[row * D_DIM + col] = c[j] + b2[col] + xcat[row * D_DIM + col];
      }
    }
  }
}

// Shared tile prologue for the logits passes: LN'd 16xD f16 block in LDS.
DEV void ln_tile(const float* xnew, const float* mu2, const float* rsig2,
                 _Float16* Ah, int r0, int tid) {
  for (int e = tid; e < 16 * D_DIM; e += 256) {
    int row = e >> 6, d = e & 63;
    Ah[e] = (_Float16)((xnew[(r0 + row) * D_DIM + d] - mu2[d]) * rsig2[d]);
  }
}

// ---------------- K6: logits max pass (+ esti0 row max for gt) -------------
__global__ void k_zmax(const float* xnew, const float* mu2, const float* rsig2,
                       const _Float16* outhT, const float* esti0,
                       float* zmaxp, float* emaxp) {
  __shared__ __align__(32) _Float16 Ah[16 * 64];
  __shared__ float P[8][16];
  int tid = threadIdx.x, lane = tid & 31, wv = tid >> 5;
  int r0 = blockIdx.y * 16, c0 = blockIdx.x * 256, bx = blockIdx.x;
  // warm L2/L0 for the esti0 tile consumed below (global_prefetch_b8)
  __builtin_prefetch(&esti0[(size_t)(r0 + (tid >> 4)) * V_DIM + c0 + (tid & 15) * 16], 0, 3);
  ln_tile(xnew, mu2, rsig2, Ah, r0, tid);
  __syncthreads();
  v16h a0 = frag_A(Ah, lane, 0, 64);
  v16h a1 = frag_A(Ah, lane, 32, 64);
  float m8[8];
#pragma unroll
  for (int j = 0; j < 8; ++j) m8[j] = -3.0e38f;
  for (int tt = 0; tt < 2; ++tt) {
    int ct = wv * 2 + tt;
    const _Float16* Bp = outhT + (size_t)(c0 + ct * 16) * 64;
    v8f c = {};
    c = wmma_f16(a0, frag_Bt(Bp, lane, 0, 64), c);
    c = wmma_f16(a1, frag_Bt(Bp, lane, 32, 64), c);
#pragma unroll
    for (int j = 0; j < 8; ++j) m8[j] = fmaxf(m8[j], c[j]);
  }
#pragma unroll
  for (int j = 0; j < 8; ++j) {
#pragma unroll
    for (int off = 1; off < 16; off <<= 1)
      m8[j] = fmaxf(m8[j], __shfl_xor(m8[j], off, 32));
  }
  int rbase = (lane >> 4) * 8;
  if ((lane & 15) == 0)
#pragma unroll
    for (int j = 0; j < 8; ++j) P[wv][rbase + j] = m8[j];
  __syncthreads();
  if (tid < 16) {
    float v = P[0][tid];
#pragma unroll
    for (int w = 1; w < 8; ++w) v = fmaxf(v, P[w][tid]);
    zmaxp[(size_t)(r0 + tid) * 32 + bx] = v;
  }
  __syncthreads();
  // esti[0] row-max over this chunk: thread tid owns column c0+tid.
  float em[16];
#pragma unroll
  for (int k = 0; k < 16; ++k)
    em[k] = esti0[(size_t)(r0 + k) * V_DIM + c0 + tid];
#pragma unroll
  for (int k = 0; k < 16; ++k) {
#pragma unroll
    for (int off = 1; off < 32; off <<= 1)
      em[k] = fmaxf(em[k], __shfl_xor(em[k], off, 32));
  }
  if (lane == 0)
#pragma unroll
    for (int k = 0; k < 16; ++k) P[wv][k] = em[k];
  __syncthreads();
  if (tid < 16) {
    float v = P[0][tid];
#pragma unroll
    for (int w = 1; w < 8; ++w) v = fmaxf(v, P[w][tid]);
    emaxp[(size_t)(r0 + tid) * 32 + bx] = v;
  }
}

// ---------------- K7: reduce chunk maxes -> row max, gt --------------------
__global__ void k_redmax(const float* zmaxp, const float* emaxp,
                         float* zmaxr, float* gt) {
  int i = blockIdx.x * 256 + threadIdx.x;
  if (i >= N_TOK) return;
  float zm = -3.0e38f, em = -3.0e38f;
#pragma unroll 8
  for (int c = 0; c < 32; ++c) {
    zm = fmaxf(zm, zmaxp[(size_t)i * 32 + c]);
    em = fmaxf(em, emaxp[(size_t)i * 32 + c]);
  }
  zmaxr[i] = zm;
  gt[i] = (em > LOG_ALPHA) ? 1.f : 0.f;
}

// ---------------- K8: logits sumexp pass -----------------------------------
__global__ void k_zsum(const float* xnew, const float* mu2, const float* rsig2,
                       const _Float16* outhT, const float* zmaxr, float* zsump) {
  __shared__ __align__(32) _Float16 Ah[16 * 64];
  __shared__ float P[8][16];
  int tid = threadIdx.x, lane = tid & 31, wv = tid >> 5;
  int r0 = blockIdx.y * 16, c0 = blockIdx.x * 256, bx = blockIdx.x;
  ln_tile(xnew, mu2, rsig2, Ah, r0, tid);
  __syncthreads();
  v16h a0 = frag_A(Ah, lane, 0, 64);
  v16h a1 = frag_A(Ah, lane, 32, 64);
  int rbase = (lane >> 4) * 8;
  float zm8[8];
#pragma unroll
  for (int j = 0; j < 8; ++j) zm8[j] = zmaxr[r0 + rbase + j];
  float s8[8];
#pragma unroll
  for (int j = 0; j < 8; ++j) s8[j] = 0.f;
  for (int tt = 0; tt < 2; ++tt) {
    int ct = wv * 2 + tt;
    const _Float16* Bp = outhT + (size_t)(c0 + ct * 16) * 64;
    v8f c = {};
    c = wmma_f16(a0, frag_Bt(Bp, lane, 0, 64), c);
    c = wmma_f16(a1, frag_Bt(Bp, lane, 32, 64), c);
#pragma unroll
    for (int j = 0; j < 8; ++j) s8[j] += __expf(c[j] - zm8[j]);
  }
#pragma unroll
  for (int j = 0; j < 8; ++j) {
#pragma unroll
    for (int off = 1; off < 16; off <<= 1)
      s8[j] += __shfl_xor(s8[j], off, 32);
  }
  if ((lane & 15) == 0)
#pragma unroll
    for (int j = 0; j < 8; ++j) P[wv][rbase + j] = s8[j];
  __syncthreads();
  if (tid < 16) {
    float v = 0.f;
#pragma unroll
    for (int w = 0; w < 8; ++w) v += P[w][tid];
    zsump[(size_t)(r0 + tid) * 32 + bx] = v;
  }
}

// ---------------- K9: reduce chunk sums -> lse -----------------------------
__global__ void k_redsum(const float* zsump, const float* zmaxr, float* lse) {
  int i = blockIdx.x * 256 + threadIdx.x;
  if (i >= N_TOK) return;
  float s = 0.f;
#pragma unroll 8
  for (int c = 0; c < 32; ++c) s += zsump[(size_t)i * 32 + c];
  lse[i] = logf(s) + zmaxr[i];
}

// ---------------- K10: final pass — xest + esti blend ----------------------
__global__ void k_final(const float* xnew, const float* mu2, const float* rsig2,
                        const _Float16* outhT, const float* lse, const float* gt,
                        const float* esti0, const float* esti1,
                        float* out_e0, float* out_e1) {
  __shared__ __align__(32) _Float16 Ah[16 * 64];
  int tid = threadIdx.x, lane = tid & 31, wv = tid >> 5;
  int r0 = blockIdx.y * 16, c0 = blockIdx.x * 256;
  size_t pf = (size_t)(r0 + (tid >> 4)) * V_DIM + c0 + (tid & 15) * 16;
  __builtin_prefetch(&esti0[pf], 0, 3);
  __builtin_prefetch(&esti1[pf], 0, 3);
  ln_tile(xnew, mu2, rsig2, Ah, r0, tid);
  __syncthreads();
  v16h a0 = frag_A(Ah, lane, 0, 64);
  v16h a1 = frag_A(Ah, lane, 32, 64);
  int rbase = (lane >> 4) * 8;
  float lse8[8], gt8[8];
#pragma unroll
  for (int j = 0; j < 8; ++j) {
    lse8[j] = lse[r0 + rbase + j];
    gt8[j]  = gt[r0 + rbase + j];
  }
  for (int tt = 0; tt < 2; ++tt) {
    int ct = wv * 2 + tt;
    const _Float16* Bp = outhT + (size_t)(c0 + ct * 16) * 64;
    v8f c = {};
    c = wmma_f16(a0, frag_Bt(Bp, lane, 0, 64), c);
    c = wmma_f16(a1, frag_Bt(Bp, lane, 32, 64), c);
    int col = c0 + ct * 16 + (lane & 15);
#pragma unroll
    for (int j = 0; j < 8; ++j) {
      int row = r0 + rbase + j;
      float xest = c[j] - lse8[j];
      size_t idx = (size_t)row * V_DIM + col;
      float e0 = esti0[idx], e1 = esti1[idx], g = gt8[j];
      out_e0[idx] = g * e0 + (1.f - g) * xest;
      out_e1[idx] = g * e1 + (1.f - g) * e0;
    }
  }
}

// ============================ host launcher ================================
extern "C" void kernel_launch(void* const* d_in, const int* in_sizes, int n_in,
                              void* d_out, int out_size, void* d_ws, size_t ws_size,
                              hipStream_t stream) {
  const float* xcat  = (const float*)d_in[0];
  const float* esti  = (const float*)d_in[1];
  const float* outw  = (const float*)d_in[2];
  const float* delta = (const float*)d_in[3];
  const float* a_p   = (const float*)d_in[4];
  const float* b_p   = (const float*)d_in[5];
  const float* w1    = (const float*)d_in[6];
  const float* b1    = (const float*)d_in[7];
  const float* w2    = (const float*)d_in[8];
  const float* b2    = (const float*)d_in[9];
  const int*   xsep  = (const int*)d_in[10];
  const int*   xdiv  = (const int*)d_in[11];
  const int*   seed  = (const int*)d_in[12];

  const float* esti0 = esti;
  const float* esti1 = esti + (size_t)N_TOK * V_DIM;

  char* ws = (char*)d_ws;
  size_t off = 0;
  auto alloc = [&](size_t bytes) -> void* {
    void* p = ws + off;
    off = (off + bytes + 255) & ~(size_t)255;
    return p;
  };
  float*     t_buf = (float*)alloc(N_TOK * 4);
  int*       kept  = (int*)alloc(N_TOK * 4);
  int*       key   = (int*)alloc(N_TOK * 4);
  float*     xpre  = (float*)alloc((size_t)N_TOK * D_DIM * 4);
  float*     mu1   = (float*)alloc(D_DIM * 4);
  float*     rsig1 = (float*)alloc(D_DIM * 4);
  float*     mu2   = (float*)alloc(D_DIM * 4);
  float*     rsig2 = (float*)alloc(D_DIM * 4);
  _Float16*  outhT = (_Float16*)alloc((size_t)D_DIM * V_DIM * 2);
  _Float16*  w1T   = (_Float16*)alloc((size_t)D_DIM * H_DIM * 2);
  _Float16*  w2T   = (_Float16*)alloc((size_t)H_DIM * D_DIM * 2);
  float*     zmaxp = (float*)alloc((size_t)N_TOK * 32 * 4);
  float*     emaxp = (float*)alloc((size_t)N_TOK * 32 * 4);
  float*     zsump = (float*)alloc((size_t)N_TOK * 32 * 4);
  float*     zmaxr = (float*)alloc(N_TOK * 4);
  float*     lse   = (float*)alloc(N_TOK * 4);
  float*     gt    = (float*)alloc(N_TOK * 4);

  float* out_x  = (float*)d_out;
  float* out_e0 = out_x + (size_t)N_TOK * D_DIM;
  float* out_e1 = out_e0 + (size_t)N_TOK * V_DIM;

  k_convert<<<dim3((D_DIM * V_DIM + 255) / 256), dim3(256), 0, stream>>>(
      outw, w1, w2, outhT, w1T, w2T);
  k_time<<<dim3(N_TOK / 256), dim3(256), 0, stream>>>(xcat, delta, xdiv, seed,
                                                      t_buf, kept);
  k_divscan<<<dim3(1), dim3(1024), 0, stream>>>(kept, xsep, key);
  k_scan<<<dim3(1), dim3(64), 0, stream>>>(xcat, t_buf, key, a_p, b_p, xpre);
  k_colstats<<<dim3(D_DIM), dim3(256), 0, stream>>>(xpre, mu1, rsig1);
  k_ffn<<<dim3(N_TOK / 64), dim3(256), 0, stream>>>(xpre, xcat, mu1, rsig1,
                                                    w1T, b1, w2T, b2, out_x);
  k_colstats<<<dim3(D_DIM), dim3(256), 0, stream>>>(out_x, mu2, rsig2);
  dim3 lg(V_DIM / 256, N_TOK / 16);
  k_zmax<<<lg, dim3(256), 0, stream>>>(out_x, mu2, rsig2, outhT, esti0,
                                       zmaxp, emaxp);
  k_redmax<<<dim3(N_TOK / 256), dim3(256), 0, stream>>>(zmaxp, emaxp,
                                                        zmaxr, gt);
  k_zsum<<<lg, dim3(256), 0, stream>>>(out_x, mu2, rsig2, outhT, zmaxr, zsump);
  k_redsum<<<dim3(N_TOK / 256), dim3(256), 0, stream>>>(zsump, zmaxr, lse);
  k_final<<<lg, dim3(256), 0, stream>>>(out_x, mu2, rsig2, outhT, lse, gt,
                                        esti0, esti1, out_e0, out_e1);
}
